// MaskHeadClip_11381663334727
// MI455X (gfx1250) — matrix operand order repaired and checked by reading, not compiled, not run on hardware
//
#include <hip/hip_runtime.h>

// ---------------- problem constants (match reference) ----------------
constexpr int B = 4, Q = 100, D = 128;
constexpr int S = 5000, N = 200000;
constexpr int N1 = 100000, N2 = 50000;
constexpr int CLS = 200, CLIP = 768;
constexpr float ANNEAL = 0.07f;
constexpr float LN_EPS = 1e-5f;

constexpr int SP = 5008;   // S padded to multiple of 16
constexpr int QP = 112;    // Q padded to multiple of 16
constexpr int ST = SP / 16;  // 313 tiles in S
constexpr int QT = QP / 16;  // 7 tiles in Q

// ---------------- workspace layout (bytes) ----------------
constexpr size_t X_OFF    = 0;                                   // [B*Q, D] f32
constexpr size_t H1_OFF   = X_OFF   + (size_t)B*Q*D*4;           // [B*Q, D] f32
constexpr size_t G1_OFF   = H1_OFF  + (size_t)B*Q*D*4;           // [B*Q, D] f32
constexpr size_t QF_OFF   = G1_OFF  + (size_t)B*Q*D*4;           // [B*Q, CLIP] f32
constexpr size_t TF_OFF   = QF_OFF  + (size_t)B*Q*CLIP*4;        // [CLS, CLIP] f32
constexpr size_t ZERO_OFF = TF_OFF  + (size_t)CLS*CLIP*4;        // zeroed region starts here
constexpr size_t MSBF_OFF = ZERO_OFF;                            // [B, SP, D] bf16
constexpr size_t MEBF_OFF = MSBF_OFF + (size_t)B*SP*D*2;         // [B, QP, D] bf16
constexpr size_t P1S_OFF  = MEBF_OFF + (size_t)B*QP*D*2;         // [B, N1, Q] f32
constexpr size_t P1C_OFF  = P1S_OFF  + (size_t)B*N1*Q*4;         // [B, N1] f32
constexpr size_t P2S_OFF  = P1C_OFF  + (size_t)B*N1*4;           // [B, N2, Q] f32
constexpr size_t P2C_OFF  = P2S_OFF  + (size_t)B*N2*Q*4;         // [B, N2] f32
constexpr size_t WS_END   = P2C_OFF  + (size_t)B*N2*4;

// ---------------- WMMA types ----------------
typedef __bf16 v16bf __attribute__((ext_vector_type(16)));
typedef float  v8f   __attribute__((ext_vector_type(8)));
struct alignas(16) B8 { __bf16 h[8]; };

// ---------------- kernel 1: layernorm ----------------
__global__ void k_layernorm(const float* __restrict__ in, const float* __restrict__ g,
                            const float* __restrict__ bb, float* __restrict__ x) {
  __shared__ float s1[128], s2[128];
  const int row = blockIdx.x, t = threadIdx.x;
  float v = in[(size_t)row * D + t];
  s1[t] = v; s2[t] = v * v;
  __syncthreads();
  for (int off = 64; off > 0; off >>= 1) {
    if (t < off) { s1[t] += s1[t + off]; s2[t] += s2[t + off]; }
    __syncthreads();
  }
  const float mu  = s1[0] * (1.0f / D);
  const float var = s2[0] * (1.0f / D) - mu * mu;
  const float inv = rsqrtf(var + LN_EPS);
  x[(size_t)row * D + t] = (v - mu) * inv * g[t] + bb[t];
}

// ---------------- kernel 2: h1 = relu(x@W1+b1), g1 = relu(x@C1+c1) ----------------
__global__ void k_mlp1(const float* __restrict__ x,
                       const float* __restrict__ W1, const float* __restrict__ b1,
                       const float* __restrict__ C1, const float* __restrict__ c1,
                       float* __restrict__ h1, float* __restrict__ g1) {
  __shared__ float xr[128];
  const int row = blockIdx.x, j = threadIdx.x;
  xr[j] = x[(size_t)row * D + j];
  __syncthreads();
  float a1 = b1[j], a2 = c1[j];
  #pragma unroll 8
  for (int k = 0; k < D; ++k) {
    const float xv = xr[k];
    a1 = fmaf(xv, W1[k * D + j], a1);
    a2 = fmaf(xv, C1[k * D + j], a2);
  }
  h1[(size_t)row * D + j] = fmaxf(a1, 0.0f);
  g1[(size_t)row * D + j] = fmaxf(a2, 0.0f);
}

// ---------------- kernel 3: mask_embed = h1@W2+b2, stored as padded bf16 ----------------
__global__ void k_mlp2(const float* __restrict__ h1,
                       const float* __restrict__ W2, const float* __restrict__ b2,
                       __bf16* __restrict__ mebf) {
  __shared__ float hr[128];
  const int row = blockIdx.x, j = threadIdx.x;
  const int b = row / Q, q = row % Q;
  hr[j] = h1[(size_t)row * D + j];
  __syncthreads();
  float a = b2[j];
  #pragma unroll 8
  for (int k = 0; k < D; ++k) a = fmaf(hr[k], W2[k * D + j], a);
  mebf[((size_t)b * QP + q) * D + j] = (__bf16)a;
}

// ---------------- kernel 4: qf = g1@C2+c2  [B*Q, 768] ----------------
__global__ void k_qf(const float* __restrict__ g1,
                     const float* __restrict__ C2, const float* __restrict__ c2,
                     float* __restrict__ qf) {
  __shared__ float gr[128];
  const int row = blockIdx.x, t = threadIdx.x;
  if (t < D) gr[t] = g1[(size_t)row * D + t];
  __syncthreads();
  for (int c = t; c < CLIP; c += blockDim.x) {
    float a = c2[c];
    #pragma unroll 8
    for (int k = 0; k < D; ++k) a = fmaf(gr[k], C2[(size_t)k * CLIP + c], a);
    qf[(size_t)row * CLIP + c] = a;
  }
}

// ---------------- kernel 5: row-wise l2norm (qf in place; clip_feat -> tf) ----------------
__global__ void k_l2norm(float* __restrict__ qf, const float* __restrict__ clip,
                         float* __restrict__ tf) {
  __shared__ float ss[256];
  const int row = blockIdx.x, t = threadIdx.x;
  const float* src; float* dst;
  if (row < B * Q) { src = qf + (size_t)row * CLIP; dst = qf + (size_t)row * CLIP; }
  else { src = clip + (size_t)(row - B * Q) * CLIP; dst = tf + (size_t)(row - B * Q) * CLIP; }
  float acc = 0.0f;
  for (int k = t; k < CLIP; k += 256) { const float v = src[k]; acc = fmaf(v, v, acc); }
  ss[t] = acc;
  __syncthreads();
  for (int off = 128; off > 0; off >>= 1) {
    if (t < off) ss[t] += ss[t + off];
    __syncthreads();
  }
  const float scale = 1.0f / fmaxf(sqrtf(ss[0]), 1e-12f);
  for (int k = t; k < CLIP; k += 256) dst[k] = src[k] * scale;
}

// ---------------- kernel 6: class head ----------------
__global__ void k_class(const float* __restrict__ qf, const float* __restrict__ tf,
                        float* __restrict__ out_class) {
  __shared__ float qr[CLIP];
  const int row = blockIdx.x, t = threadIdx.x;
  for (int k = t; k < CLIP; k += 256) qr[k] = qf[(size_t)row * CLIP + k];
  __syncthreads();
  if (t < CLS) {
    float a = 0.0f;
    const float* tr = tf + (size_t)t * CLIP;
    #pragma unroll 8
    for (int k = 0; k < CLIP; ++k) a = fmaf(qr[k], tr[k], a);
    float r = a * (1.0f / ANNEAL);
    if (t == 0 || t == 2) r = -__builtin_inff();
    out_class[(size_t)row * CLS + t] = r;
  }
}

// ---------------- kernel 7: f32 -> padded bf16 copy of mask_segments ----------------
__global__ void k_convseg(const float* __restrict__ ms, __bf16* __restrict__ msbf) {
  const size_t idx = (size_t)blockIdx.x * blockDim.x + threadIdx.x;
  const int d = (int)(idx % D);
  const int s = (int)((idx / D) % S);
  const int b = (int)(idx / ((size_t)D * S));
  msbf[((size_t)b * SP + s) * D + d] = (__bf16)ms[idx];
}

// ---------------- kernel 8: output_segments via bf16 WMMA ----------------
// out[b, s, q] = dot(mask_segments[b,s,:], mask_embed[b,q,:]), K = 128 = 4 x K32
__global__ void k_wmma_seg(const __bf16* __restrict__ msbf,
                           const __bf16* __restrict__ mebf,
                           float* __restrict__ outseg) {
  int tile = blockIdx.x;
  const int qt = tile % QT; tile /= QT;
  const int st = tile % ST; const int b = tile / ST;
  const int L = threadIdx.x;      // 0..31
  const int half = L >> 4;        // lane half-group
  const int l16  = L & 15;

  const __bf16* arow = msbf + ((size_t)b * SP + st * 16 + l16) * D;  // A: row M=l16
  const __bf16* brow = mebf + ((size_t)b * QP + qt * 16 + l16) * D;  // B: col N=l16

  v8f c = {};
  #pragma unroll
  for (int kc = 0; kc < 4; ++kc) {
    const int base = kc * 32;
    // A fragment (16x32 bf16): elem e -> K = base + (e<8?0:16) + half*8 + (e&7)
    const B8 alo = *(const B8*)(arow + base + half * 8);
    const B8 ahi = *(const B8*)(arow + base + 16 + half * 8);
    // B fragment (32x16 bf16): elem e -> K = base + (e<8?0:16) + half*8 + (e&7)
    const B8 blo = *(const B8*)(brow + base + half * 8);
    const B8 bhi = *(const B8*)(brow + base + 16 + half * 8);
    v16bf av, bv;
    #pragma unroll
    for (int e = 0; e < 8; ++e) {
      av[e] = alo.h[e]; av[e + 8] = ahi.h[e];
      bv[e] = blo.h[e]; bv[e + 8] = bhi.h[e];
    }
    c = __builtin_amdgcn_wmma_f32_16x16x32_bf16(false, av, false, bv,
                                                (short)0, c, false, false);
  }
  // C/D layout: VGPR r -> M = r + half*8, N = l16
  const int nq = qt * 16 + l16;
  #pragma unroll
  for (int r = 0; r < 8; ++r) {
    const int m = st * 16 + half * 8 + r;
    if (m < S && nq < Q) outseg[((size_t)b * S + m) * Q + nq] = c[r];
  }
}

// ---------------- kernel 9: fused gather + pool1 accumulate ----------------
__global__ void k_pool1(const float* __restrict__ outseg, const int* __restrict__ p2s,
                        const int* __restrict__ ids1, float* __restrict__ p1s,
                        float* __restrict__ p1c) {
  const size_t idx = (size_t)blockIdx.x * blockDim.x + threadIdx.x;
  const int q = (int)(idx % Q);
  const int n = (int)((idx / Q) % N);
  const int b = (int)(idx / ((size_t)Q * N));
  const int s = p2s[(size_t)b * N + n];
  const int i = ids1[(size_t)b * N + n];
  const float v = outseg[((size_t)b * S + s) * Q + q];
  atomicAdd(&p1s[((size_t)b * N1 + i) * Q + q], v);
  if (q == 0) atomicAdd(&p1c[(size_t)b * N1 + i], 1.0f);
}

// ---------------- kernel 10: pool1 mean -> pool2 accumulate ----------------
__global__ void k_pool2(const float* __restrict__ p1s, const float* __restrict__ p1c,
                        const int* __restrict__ ids2, float* __restrict__ p2s,
                        float* __restrict__ p2c) {
  const size_t idx = (size_t)blockIdx.x * blockDim.x + threadIdx.x;
  const int q = (int)(idx % Q);
  const int i = (int)((idx / Q) % N1);
  const int b = (int)(idx / ((size_t)Q * N1));
  const float cnt = fmaxf(p1c[(size_t)b * N1 + i], 1.0f);
  const float mean = p1s[((size_t)b * N1 + i) * Q + q] / cnt;
  const int j = ids2[(size_t)b * N1 + i];
  atomicAdd(&p2s[((size_t)b * N2 + j) * Q + q], mean);
  if (q == 0) atomicAdd(&p2c[(size_t)b * N2 + j], 1.0f);
}

// ---------------- kernel 11: attn_mask = (pool2 mean < 0)  [sigmoid(x)<0.5] ----------------
__global__ void k_attn(const float* __restrict__ p2s, const float* __restrict__ p2c,
                       float* __restrict__ out_attn) {
  const size_t idx = (size_t)blockIdx.x * blockDim.x + threadIdx.x;
  const int q = (int)(idx % Q);
  const int j = (int)((idx / Q) % N2);
  const int b = (int)(idx / ((size_t)Q * N2));
  const float cnt = fmaxf(p2c[(size_t)b * N2 + j], 1.0f);
  const float mean = p2s[((size_t)b * N2 + j) * Q + q] / cnt;
  out_attn[idx] = (mean < 0.0f) ? 1.0f : 0.0f;
}

// ---------------- launcher ----------------
extern "C" void kernel_launch(void* const* d_in, const int* in_sizes, int n_in,
                              void* d_out, int out_size, void* d_ws, size_t ws_size,
                              hipStream_t stream) {
  (void)in_sizes; (void)n_in; (void)out_size; (void)ws_size;
  const float* query_feat    = (const float*)d_in[0];
  const float* mask_segments = (const float*)d_in[1];
  const int*   p2s           = (const int*)d_in[2];
  const int*   ids1          = (const int*)d_in[3];
  const int*   ids2          = (const int*)d_in[4];
  const float* ln_g          = (const float*)d_in[5];
  const float* ln_b          = (const float*)d_in[6];
  const float* W1 = (const float*)d_in[7];
  const float* b1 = (const float*)d_in[8];
  const float* W2 = (const float*)d_in[9];
  const float* b2 = (const float*)d_in[10];
  const float* C1 = (const float*)d_in[11];
  const float* c1 = (const float*)d_in[12];
  const float* C2 = (const float*)d_in[13];
  const float* c2 = (const float*)d_in[14];
  const float* clip_feat = (const float*)d_in[15];

  char* ws = (char*)d_ws;
  float*  x    = (float*)(ws + X_OFF);
  float*  h1   = (float*)(ws + H1_OFF);
  float*  g1   = (float*)(ws + G1_OFF);
  float*  qf   = (float*)(ws + QF_OFF);
  float*  tf   = (float*)(ws + TF_OFF);
  __bf16* msbf = (__bf16*)(ws + MSBF_OFF);
  __bf16* mebf = (__bf16*)(ws + MEBF_OFF);
  float*  p1s  = (float*)(ws + P1S_OFF);
  float*  p1c  = (float*)(ws + P1C_OFF);
  float*  p2s2 = (float*)(ws + P2S_OFF);
  float*  p2c  = (float*)(ws + P2C_OFF);

  float* out_class = (float*)d_out;                       // [B, Q, CLS]
  float* out_seg   = out_class + (size_t)B * Q * CLS;     // [B, S, Q]
  float* out_attn  = out_seg   + (size_t)B * S * Q;       // [B, N2, Q] (bool as 0/1)

  // zero pool accumulators + bf16 pad rows every call (deterministic, capturable)
  hipMemsetAsync(ws + ZERO_OFF, 0, WS_END - ZERO_OFF, stream);

  k_layernorm<<<B * Q, 128, 0, stream>>>(query_feat, ln_g, ln_b, x);
  k_mlp1<<<B * Q, 128, 0, stream>>>(x, W1, b1, C1, c1, h1, g1);
  k_mlp2<<<B * Q, 128, 0, stream>>>(h1, W2, b2, mebf);
  k_qf<<<B * Q, 256, 0, stream>>>(g1, C2, c2, qf);
  k_l2norm<<<B * Q + CLS, 256, 0, stream>>>(qf, clip_feat, tf);
  k_class<<<B * Q, 256, 0, stream>>>(qf, tf, out_class);
  k_convseg<<<(B * S * D) / 256, 256, 0, stream>>>(mask_segments, msbf);
  k_wmma_seg<<<B * ST * QT, 32, 0, stream>>>(msbf, mebf, out_seg);
  k_pool1<<<(int)(((size_t)B * N * Q) / 256), 256, 0, stream>>>(out_seg, p2s, ids1, p1s, p1c);
  k_pool2<<<(int)(((size_t)B * N1 * Q) / 256), 256, 0, stream>>>(p1s, p1c, ids2, p2s2, p2c);
  k_attn<<<(int)(((size_t)B * N2 * Q) / 256), 256, 0, stream>>>(p2s2, p2c, out_attn);
}
